// HicLSTM3D_85615878078995
// MI455X (gfx1250) — matrix operand order repaired and checked by reading, not compiled, not run on hardware
//
#include <hip/hip_runtime.h>

// Problem sizes
#define GG 300000
#define EE 256
#define HH 512
#define TT 256
#define BB 256
#define DD 512      // 2*E
#define G4 2048     // 4*H
#define KOUT 131072 // H*T

typedef __attribute__((ext_vector_type(16))) __bf16 v16bf;
typedef __attribute__((ext_vector_type(8)))  float  v8f;
typedef unsigned int u32x4 __attribute__((ext_vector_type(4)));
typedef int          i32x4 __attribute__((ext_vector_type(4)));
typedef int          i32x8 __attribute__((ext_vector_type(8)));

union FragU { v16bf v; uint4 u[2]; };

__device__ __forceinline__ unsigned short f2bf(float f) {
  unsigned int u = __float_as_uint(f);
  u += 0x7FFFu + ((u >> 16) & 1u);       // round-to-nearest-even
  return (unsigned short)(u >> 16);
}
__device__ __forceinline__ float bf2f(unsigned short h) {
  return __uint_as_float(((unsigned int)h) << 16);
}
__device__ __forceinline__ float sigm(float x) { return 1.0f / (1.0f + __expf(-x)); }

// A fragment (16x32 bf16): lane m = lane&15, khalf = (lane>>4)*8.
// rowp points at element [m][k0+khalf] of a row-major bf16 matrix.
__device__ __forceinline__ v16bf ld_fragA(const unsigned short* rowp) {
  FragU f;
  const uint4* q = (const uint4*)rowp;
  f.u[0] = q[0];        // k = khalf .. khalf+7
  f.u[1] = q[2];        // k = khalf+16 .. khalf+23  (+32 bytes)
  return f.v;
}
// B fragment (32x16 bf16): lane n = lane&15, kb = (lane>>4)*8; k = kb..kb+7, kb+16..kb+23.
__device__ __forceinline__ v16bf ld_fragB(const unsigned short* p) {
  FragU f;
  const uint4* q = (const uint4*)p;
  f.u[0] = q[0];
  f.u[1] = q[2];
  return f.v;
}

// TDM: load a [rows x cols] bf16 tile (row-major, stride = cols elements)
// from global memory into LDS. Issued by the calling wave; completion via
// TENSORcnt. Descriptor per CDNA5 ISA (2-D, groups 2/3 = 0).
__device__ __forceinline__ void tdm_load_tile_bf16(
    unsigned int lds_off, const unsigned short* gptr, int rows, int cols) {
  unsigned long long ga = (unsigned long long)(size_t)gptr;
  u32x4 g0;
  g0[0] = 1u;                                   // count=1, user mode
  g0[1] = lds_off;                              // lds_addr (bytes)
  g0[2] = (unsigned int)ga;                     // global_addr[31:0]
  g0[3] = ((unsigned int)(ga >> 32) & 0x01FFFFFFu) | (2u << 30); // [56:32] | type=2
  i32x8 g1;
  g1[0] = (int)(1u << 16);                      // data_size=1 (2 bytes)
  g1[1] = (int)(((unsigned)cols & 0xFFFFu) << 16);       // tensor_dim0[15:0] @ bit48
  g1[2] = (int)((((unsigned)cols >> 16) & 0xFFFFu) |     // tensor_dim0[31:16]
                (((unsigned)rows & 0xFFFFu) << 16));      // tensor_dim1[15:0] @ bit80
  g1[3] = (int)((((unsigned)rows >> 16) & 0xFFFFu) |     // tensor_dim1[31:16]
                (((unsigned)cols & 0xFFFFu) << 16));      // tile_dim0 @ bit112
  g1[4] = (int)((unsigned)rows & 0xFFFFu);               // tile_dim1 (tile_dim2=0)
  g1[5] = (int)cols;                                      // tensor_dim0_stride[31:0]
  g1[6] = 0;                                              // stride hi | dim1_stride lo
  g1[7] = 0;
  i32x4 z4 = {0, 0, 0, 0};
#if defined(__clang_major__) && (__clang_major__ >= 23)
  i32x8 z8 = {0, 0, 0, 0, 0, 0, 0, 0};
  __builtin_amdgcn_tensor_load_to_lds(g0, g1, z4, z4, z8, 0);
#else
  __builtin_amdgcn_tensor_load_to_lds(g0, g1, z4, z4, 0);
#endif
  __builtin_amdgcn_s_wait_tensorcnt(0);
}

// ---------------------------------------------------------------------------
// Kernel 0: convert weights to bf16 scratch, init h (bf16) and c (f32) state.
// ---------------------------------------------------------------------------
__global__ __launch_bounds__(256) void k_prep(
    const float* __restrict__ w_ih, const float* __restrict__ w_hh,
    const float* __restrict__ w_out, const float* __restrict__ h0,
    const float* __restrict__ c0,
    unsigned short* __restrict__ wih_bf, unsigned short* __restrict__ whh_bf,
    unsigned short* __restrict__ wout_bf, unsigned short* __restrict__ hbf,
    float* __restrict__ cbuf) {
  size_t i = (size_t)blockIdx.x * 256 + threadIdx.x;
  size_t stride = (size_t)gridDim.x * 256;
  for (size_t e = i; e < (size_t)G4 * DD; e += stride) {
    wih_bf[e] = f2bf(w_ih[e]);
    whh_bf[e] = f2bf(w_hh[e]);
  }
  for (size_t e = i; e < (size_t)TT * KOUT; e += stride)
    wout_bf[e] = f2bf(w_out[e]);
  for (size_t e = i; e < (size_t)BB * HH; e += stride) {
    hbf[e] = f2bf(h0[e]);
    cbuf[e] = c0[e];
  }
}

// ---------------------------------------------------------------------------
// Kernel 1: xg = LN(embed[ids] @ w_ih^T) * g_x + b_x   -> bf16 (B*T, 4H)
// ---------------------------------------------------------------------------
__global__ __launch_bounds__(256) void k_xgate(
    const int* __restrict__ ids, const float* __restrict__ embed,
    const unsigned short* __restrict__ wih_bf,
    const float* __restrict__ gx, const float* __restrict__ bx,
    unsigned short* __restrict__ xg) {
  __shared__ unsigned short xlds[16 * DD];   // 16 KB (gathered x tile, bf16)
  __shared__ unsigned short gates[16 * G4];  // 64 KB (pre-LN gates, bf16)
  __shared__ float redS[256], redQ[256];
  __shared__ int lid[32];

  const int tid = threadIdx.x, lane = tid & 31, wave = tid >> 5;
  const int wg = blockIdx.x;  // tile of 16 consecutive bt = b*T + t rows

  if (tid < 32) lid[tid] = ids[(size_t)wg * 32 + tid];
  __syncthreads();

  // Gather + f32->bf16 into LDS: row m = concat(embed[id0], embed[id1])
  for (int e = tid; e < 16 * DD; e += 256) {
    int m = e >> 9, col = e & (DD - 1);
    int id = (col < EE) ? lid[m * 2] : lid[m * 2 + 1];
    xlds[e] = f2bf(embed[(size_t)id * EE + (col & (EE - 1))]);
  }
  __syncthreads();

  // GEMM: 16x2048 += 16x512 * 512x2048 (B = w_ih, N-major in memory)
  {
    const int mA = lane & 15, khA = (lane >> 4) * 8;
    const int nB = lane & 15, kbB = (lane >> 4) * 8;
    for (int nt = wave; nt < G4 / 16; nt += 8) {
      v8f acc = {};
      const unsigned short* wrow = wih_bf + (size_t)(nt * 16 + nB) * DD + kbB;
      const unsigned short* arow = xlds + mA * DD + khA;
#pragma unroll 4
      for (int ks = 0; ks < DD / 32; ++ks) {
        v16bf a = ld_fragA(arow + ks * 32);
        v16bf b = ld_fragB(wrow + ks * 32);
        acc = __builtin_amdgcn_wmma_f32_16x16x32_bf16(
            false, a, false, b, (short)0, acc, false, false);
      }
      int col = nt * 16 + (lane & 15);
#pragma unroll
      for (int r = 0; r < 8; ++r) {
        int m = r + (lane >> 4) * 8;
        gates[m * G4 + col] = f2bf(acc[r]);
      }
    }
  }
  __syncthreads();

  // LayerNorm over 2048 per row, scale/shift, write bf16 xg
  const int row = tid >> 4, sub = tid & 15;
  float s = 0.f, q = 0.f;
  for (int c = sub * 128; c < sub * 128 + 128; ++c) {
    float v = bf2f(gates[row * G4 + c]);
    s += v; q += v * v;
  }
  redS[tid] = s; redQ[tid] = q;
  __syncthreads();
  float S = 0.f, Q = 0.f;
#pragma unroll
  for (int j = 0; j < 16; ++j) { S += redS[row * 16 + j]; Q += redQ[row * 16 + j]; }
  float mean = S * (1.0f / G4);
  float rstd = rsqrtf(Q * (1.0f / G4) - mean * mean + 1e-5f);
  size_t obase = ((size_t)wg * 16 + row) * G4;
  for (int c = sub * 128; c < sub * 128 + 128; ++c) {
    float v = (bf2f(gates[row * G4 + c]) - mean) * rstd * gx[c] + bx[c];
    xg[obase + c] = f2bf(v);
  }
}

// ---------------------------------------------------------------------------
// Kernel 2: one LSTM time step (launched T times, stream-ordered).
// h tile staged via Tensor Data Mover (TENSORcnt), WMMA from LDS.
// ---------------------------------------------------------------------------
__global__ __launch_bounds__(256) void k_step(
    int t, const unsigned short* __restrict__ xg,
    const unsigned short* __restrict__ whh_bf, const float* __restrict__ bias,
    const float* __restrict__ gh, const float* __restrict__ bh,
    const float* __restrict__ gc, const float* __restrict__ bc,
    unsigned short* __restrict__ hbf, float* __restrict__ cbuf,
    unsigned short* __restrict__ hs) {
  __shared__ unsigned short hlds[16 * HH];   // 16 KB (h tile via TDM)
  __shared__ unsigned short gates[16 * G4];  // 64 KB
  __shared__ float clds[16 * HH];            // 32 KB
  __shared__ float redS[256], redQ[256];

  const int tid = threadIdx.x, lane = tid & 31, wave = tid >> 5;
  const int b0 = blockIdx.x * 16;

  // Stage h(16x512 bf16) global -> LDS with one TDM descriptor (wave 0 only;
  // TDM ignores EXEC, tracked by TENSORcnt).
  if (tid < 32) {
    tdm_load_tile_bf16((unsigned int)(size_t)(void*)hlds,
                       hbf + (size_t)b0 * HH, 16, HH);
  }
  __syncthreads();

  // GEMM: z = h(16x512) @ w_hh^T -> 16x2048
  {
    const int mA = lane & 15, khA = (lane >> 4) * 8;
    const int nB = lane & 15, kbB = (lane >> 4) * 8;
    const unsigned short* arow = hlds + mA * HH + khA;
    for (int nt = wave; nt < G4 / 16; nt += 8) {
      v8f acc = {};
      const unsigned short* wrow = whh_bf + (size_t)(nt * 16 + nB) * HH + kbB;
#pragma unroll 4
      for (int ks = 0; ks < HH / 32; ++ks) {
        v16bf a = ld_fragA(arow + ks * 32);
        v16bf b = ld_fragB(wrow + ks * 32);
        acc = __builtin_amdgcn_wmma_f32_16x16x32_bf16(
            false, a, false, b, (short)0, acc, false, false);
      }
      int col = nt * 16 + (lane & 15);
#pragma unroll
      for (int r = 0; r < 8; ++r) {
        int m = r + (lane >> 4) * 8;
        gates[m * G4 + col] = f2bf(acc[r]);
      }
    }
  }
  __syncthreads();

  const int row = tid >> 4, sub = tid & 15;
  const int b = b0 + row;

  // LN(z) over 2048, then gates = xg + LN(z)*g_h + b_h + bias (back to LDS)
  float s = 0.f, q = 0.f;
  for (int c = sub * 128; c < sub * 128 + 128; ++c) {
    float v = bf2f(gates[row * G4 + c]);
    s += v; q += v * v;
  }
  redS[tid] = s; redQ[tid] = q;
  __syncthreads();
  float S = 0.f, Q = 0.f;
#pragma unroll
  for (int j = 0; j < 16; ++j) { S += redS[row * 16 + j]; Q += redQ[row * 16 + j]; }
  float mean = S * (1.0f / G4);
  float rstd = rsqrtf(Q * (1.0f / G4) - mean * mean + 1e-5f);
  size_t xgb = ((size_t)b * TT + t) * G4;
  for (int c = sub * 128; c < sub * 128 + 128; ++c) {
    float z = bf2f(gates[row * G4 + c]);
    float v = bf2f(xg[xgb + c]) + (z - mean) * rstd * gh[c] + bh[c] + bias[c];
    gates[row * G4 + c] = f2bf(v);
  }
  __syncthreads();

  // c update + stats for LN(c)
  float s2 = 0.f, q2 = 0.f;
  for (int j = sub * 32; j < sub * 32 + 32; ++j) {
    float gi = bf2f(gates[row * G4 + j]);
    float gf = bf2f(gates[row * G4 + HH + j]);
    float gg = bf2f(gates[row * G4 + 2 * HH + j]);
    float co = cbuf[(size_t)b * HH + j];
    float cn = sigm(gf) * co + sigm(gi) * tanhf(gg);
    cbuf[(size_t)b * HH + j] = cn;
    clds[row * HH + j] = cn;
    s2 += cn; q2 += cn * cn;
  }
  redS[tid] = s2; redQ[tid] = q2;
  __syncthreads();
  S = 0.f; Q = 0.f;
#pragma unroll
  for (int j = 0; j < 16; ++j) { S += redS[row * 16 + j]; Q += redQ[row * 16 + j]; }
  float mc = S * (1.0f / HH);
  float rc = rsqrtf(Q * (1.0f / HH) - mc * mc + 1e-5f);

  // h = sigmoid(o) * tanh(LN(c)); write bf16 state + hs history
  for (int j = sub * 32; j < sub * 32 + 32; ++j) {
    float go = bf2f(gates[row * G4 + 3 * HH + j]);
    float cn = clds[row * HH + j];
    float hn = sigm(go) * tanhf((cn - mc) * rc * gc[j] + bc[j]);
    unsigned short hb = f2bf(hn);
    hbf[(size_t)b * HH + j] = hb;
    hs[((size_t)b * TT + t) * HH + j] = hb;
  }
}

// ---------------------------------------------------------------------------
// Kernel 3: out(256,256) = hs(256 x 131072, bf16) @ w_out^T + b_out
// ---------------------------------------------------------------------------
__global__ __launch_bounds__(256) void k_out(
    const unsigned short* __restrict__ hs, const unsigned short* __restrict__ wo,
    const float* __restrict__ bout, float* __restrict__ out) {
  const int tid = threadIdx.x, lane = tid & 31, wave = tid >> 5;
  const int m0 = blockIdx.x * 16;
  const int n0 = blockIdx.y * 128 + wave * 16;

  const int mA = lane & 15, khA = (lane >> 4) * 8;
  const int nB = lane & 15, kbB = (lane >> 4) * 8;
  const unsigned short* arow = hs + (size_t)(m0 + mA) * KOUT + khA;
  const unsigned short* brow = wo + (size_t)(n0 + nB) * KOUT + kbB;

  v8f acc = {};
  for (int k0 = 0; k0 < KOUT; k0 += 32) {
    __builtin_prefetch(brow + k0 + 512, 0, 1);  // global_prefetch_b8 on w_out stream
    v16bf a = ld_fragA(arow + k0);
    v16bf b = ld_fragB(brow + k0);
    acc = __builtin_amdgcn_wmma_f32_16x16x32_bf16(
        false, a, false, b, (short)0, acc, false, false);
  }
  int n = n0 + (lane & 15);
  float bo = bout[n];
#pragma unroll
  for (int r = 0; r < 8; ++r) {
    int m = m0 + r + (lane >> 4) * 8;
    out[(size_t)m * TT + n] = acc[r] + bo;
  }
}

// ---------------------------------------------------------------------------
extern "C" void kernel_launch(void* const* d_in, const int* in_sizes, int n_in,
                              void* d_out, int out_size, void* d_ws, size_t ws_size,
                              hipStream_t stream) {
  const int*   ids   = (const int*)d_in[0];
  const float* h0    = (const float*)d_in[1];
  const float* c0    = (const float*)d_in[2];
  const float* embed = (const float*)d_in[3];
  const float* w_ih  = (const float*)d_in[4];
  const float* w_hh  = (const float*)d_in[5];
  const float* bias  = (const float*)d_in[6];
  const float* g_x   = (const float*)d_in[7];
  const float* b_x   = (const float*)d_in[8];
  const float* g_h   = (const float*)d_in[9];
  const float* b_h   = (const float*)d_in[10];
  const float* g_c   = (const float*)d_in[11];
  const float* b_c   = (const float*)d_in[12];
  const float* w_out = (const float*)d_in[13];
  const float* b_out = (const float*)d_in[14];
  float* out = (float*)d_out;

  // Workspace carve-up (all sizes 256B-aligned)
  char* p = (char*)d_ws;
  unsigned short* wih_bf  = (unsigned short*)p; p += (size_t)G4 * DD * 2;     //   2 MB
  unsigned short* whh_bf  = (unsigned short*)p; p += (size_t)G4 * HH * 2;     //   2 MB
  unsigned short* wout_bf = (unsigned short*)p; p += (size_t)TT * KOUT * 2;   //  64 MB
  unsigned short* hbf     = (unsigned short*)p; p += (size_t)BB * HH * 2;     // 256 KB
  float*          cbuf    = (float*)p;          p += (size_t)BB * HH * 4;     // 512 KB
  unsigned short* xgbuf   = (unsigned short*)p; p += (size_t)BB * TT * G4 * 2;// 256 MB
  unsigned short* hsbuf   = (unsigned short*)p;                               //  64 MB

  k_prep<<<2048, 256, 0, stream>>>(w_ih, w_hh, w_out, h0, c0,
                                   wih_bf, whh_bf, wout_bf, hbf, cbuf);

  k_xgate<<<(BB * TT) / 16, 256, 0, stream>>>(ids, embed, wih_bf, g_x, b_x, xgbuf);

  for (int t = 0; t < TT; ++t)
    k_step<<<BB / 16, 256, 0, stream>>>(t, xgbuf, whh_bf, bias,
                                        g_h, b_h, g_c, b_c, hbf, cbuf, hsbuf);

  k_out<<<dim3(BB / 16, 2), 256, 0, stream>>>(hsbuf, wout_bf, b_out, out);
}
